// SelfAttention_76149770158154
// MI455X (gfx1250) — compile-verified, matrix operand order checked by reading, not run on hardware
//
#include <hip/hip_runtime.h>
#include <hip/hip_bf16.h>
#include <math.h>

#define UNITS 1024
#define BATCH 32
#define SEQ   2048
#define MTILE 128           // rows of hidden per workgroup (bf16 in LDS)
#define APAD  1032          // padded LDS row stride in bf16 elements (bank-conflict-free)

typedef __attribute__((ext_vector_type(16))) __bf16 bf16x16;
typedef __attribute__((ext_vector_type(8)))  __bf16 bf16x8;
typedef __attribute__((ext_vector_type(4)))  __bf16 bf16x4;
typedef __attribute__((ext_vector_type(8)))  float  f32x8;

union FragAB { bf16x16 v; bf16x8 h[2]; };

__device__ __forceinline__ __bf16 f2bf(float f) { return (__bf16)f; }

// Branchless tanh: prefer the CDNA5 hardware transcendental v_tanh_f32.
__device__ __forceinline__ float fast_tanh(float x) {
#if defined(__HIP_DEVICE_COMPILE__)
# if __has_builtin(__builtin_amdgcn_tanhf)
    return __builtin_amdgcn_tanhf(x);
# elif __has_builtin(__builtin_amdgcn_tanh_f32)
    return __builtin_amdgcn_tanh_f32(x);
# else
    const float e = __expf(2.0f * x);                       // v_exp_f32
    return 1.0f - 2.0f * __builtin_amdgcn_rcpf(e + 1.0f);   // v_rcp_f32, saturates to +/-1
# endif
#else
    return tanhf(x);
#endif
}

// ---------------------------------------------------------------------------
// U_w (K x N, f32, row-major) -> Ut (N x K, bf16) so B fragments are two
// contiguous 16B loads per lane. 32x32 LDS tile transpose.
__global__ void k_transpose_uw(const float* __restrict__ Uw, __bf16* __restrict__ Ut) {
    __shared__ float tile[32][33];
    const int bx = blockIdx.x * 32, by = blockIdx.y * 32;
    const int tx = threadIdx.x, ty = threadIdx.y;       // block (32, 8)
    #pragma unroll
    for (int i = 0; i < 32; i += 8)
        tile[ty + i][tx] = Uw[(size_t)(by + ty + i) * UNITS + bx + tx];
    __syncthreads();
    #pragma unroll
    for (int i = 0; i < 32; i += 8)
        Ut[(size_t)(bx + ty + i) * UNITS + by + tx] = f2bf(tile[tx][ty + i]);
}

// ---------------------------------------------------------------------------
// decoder = s_prev @ W_w + W_b   (32 x 1024, tiny: 33M FMA)
__global__ void k_decoder(const float* __restrict__ s_prev, const float* __restrict__ Ww,
                          const float* __restrict__ Wb, float* __restrict__ dec) {
    const int g = blockIdx.x * blockDim.x + threadIdx.x;   // 32768
    const int b = g >> 10, h = g & 1023;
    const float* sp = s_prev + (size_t)b * UNITS;
    float acc = 0.f;
    for (int k = 0; k < UNITS; ++k)
        acc = fmaf(sp[k], Ww[(size_t)k * UNITS + h], acc);
    dec[g] = acc + Wb[h];
}

// ---------------------------------------------------------------------------
// Fused: scores[b,s] = tanh(hidden[b,s,:] @ U_w + U_b + dec[b,:]) . V_w + V_b
// One WG = 128 rows of hidden in LDS (bf16), full N loop per WG so hidden is
// read from HBM exactly once. 8 waves x 16 rows (2 waves/SIMD for latency
// hiding); all waves share the same (n0,k0) B-fragment stream -> WGP$ hits.
__global__ void __launch_bounds__(256)
k_enc_scores(const float* __restrict__ hidden, const __bf16* __restrict__ Ut,
             const float* __restrict__ dec, const float* __restrict__ Ub,
             const float* __restrict__ Vw, const float* __restrict__ Vb,
             float* __restrict__ scores) {
    extern __shared__ char smem[];
    __bf16* At   = (__bf16*)smem;                                  // [MTILE][APAD]
    float*  decL = (float*)(smem + (size_t)MTILE * APAD * 2);      // [1024]
    float*  vwL  = decL + UNITS;                                   // [1024]

    const int tid  = threadIdx.x;
    const int row0 = blockIdx.x * MTILE;          // global row in [0, B*S)
    const int b    = row0 >> 11;                  // tile never crosses a batch

    // Stage hidden tile f32 -> bf16 into LDS.
    for (int idx = tid; idx < MTILE * (UNITS / 4); idx += 256) {
        const int r  = idx >> 8;                  // 256 float4 per row
        const int c4 = (idx & 255) << 2;
        const float4 v = *(const float4*)(hidden + ((size_t)(row0 + r) * UNITS + c4));
        bf16x4 p; p[0] = f2bf(v.x); p[1] = f2bf(v.y); p[2] = f2bf(v.z); p[3] = f2bf(v.w);
        *(bf16x4*)(At + (size_t)r * APAD + c4) = p;
    }
    for (int i = tid; i < UNITS; i += 256) {
        decL[i] = dec[(size_t)b * UNITS + i] + Ub[i];
        vwL[i]  = Vw[i];
    }
    __syncthreads();

    const int wave = tid >> 5;                    // 0..7 -> 16 rows each
    const int lane = tid & 31;
    const int half = lane >> 4;                   // ISA 16-bit A/B frag layout
    const int l15  = lane & 15;
    const __bf16* arow = At + (size_t)(wave * 16 + l15) * APAD;

    float acc[8];
    #pragma unroll
    for (int r = 0; r < 8; ++r) acc[r] = 0.f;

    for (int n0 = 0; n0 < UNITS; n0 += 32) {
        f32x8 c0 = {}, c1 = {};
        const __bf16* b0row = Ut + (size_t)(n0 + l15) * UNITS;
        const __bf16* b1row = Ut + (size_t)(n0 + 16 + l15) * UNITS;
        #pragma unroll 2
        for (int k0 = 0; k0 < UNITS; k0 += 32) {
            const int ko = k0 + half * 8;
            FragAB a, fb0, fb1;
            a.h[0]   = *(const bf16x8*)(arow + ko);
            a.h[1]   = *(const bf16x8*)(arow + ko + 16);
            fb0.h[0] = *(const bf16x8*)(b0row + ko);
            fb0.h[1] = *(const bf16x8*)(b0row + ko + 16);
            fb1.h[0] = *(const bf16x8*)(b1row + ko);
            fb1.h[1] = *(const bf16x8*)(b1row + ko + 16);
            c0 = __builtin_amdgcn_wmma_f32_16x16x32_bf16(false, a.v, false, fb0.v, (short)0, c0, false, false);
            c1 = __builtin_amdgcn_wmma_f32_16x16x32_bf16(false, a.v, false, fb1.v, (short)0, c1, false, false);
        }
        // Fused epilogue: tanh(enc + dec) * V_w, accumulated per row.
        const float d0 = decL[n0 + l15],      w0 = vwL[n0 + l15];
        const float d1 = decL[n0 + 16 + l15], w1 = vwL[n0 + 16 + l15];
        #pragma unroll
        for (int r = 0; r < 8; ++r)
            acc[r] += fast_tanh(c0[r] + d0) * w0 + fast_tanh(c1[r] + d1) * w1;
    }
    // Reduce across the 16 lanes of each half (row m lives in lanes sharing half).
    #pragma unroll
    for (int m = 1; m < 16; m <<= 1) {
        #pragma unroll
        for (int r = 0; r < 8; ++r)
            acc[r] += __shfl_xor(acc[r], m, 32);
    }
    if (l15 == 0) {
        const float vb = Vb[0];
        const int g0 = row0 + wave * 16 + half * 8;       // rows g0..g0+7
        #pragma unroll
        for (int r = 0; r < 8; ++r)
            scores[g0 + r] = acc[r] + vb;
    }
}

// ---------------------------------------------------------------------------
// Softmax over S per batch; writes alpha into d_out.
__global__ void k_softmax(const float* __restrict__ scores, float* __restrict__ alpha) {
    __shared__ float sc[SEQ];
    __shared__ float red[256];
    const int b = blockIdx.x, tid = threadIdx.x;
    const float* s = scores + (size_t)b * SEQ;
    float lmax = -INFINITY;
    for (int i = tid; i < SEQ; i += 256) { const float v = s[i]; sc[i] = v; lmax = fmaxf(lmax, v); }
    red[tid] = lmax; __syncthreads();
    for (int off = 128; off > 0; off >>= 1) {
        if (tid < off) red[tid] = fmaxf(red[tid], red[tid + off]);
        __syncthreads();
    }
    const float mx = red[0];
    __syncthreads();
    float lsum = 0.f;
    for (int i = tid; i < SEQ; i += 256) { const float e = __expf(sc[i] - mx); sc[i] = e; lsum += e; }
    red[tid] = lsum; __syncthreads();
    for (int off = 128; off > 0; off >>= 1) {
        if (tid < off) red[tid] += red[tid + off];
        __syncthreads();
    }
    const float inv = 1.f / red[0];
    for (int i = tid; i < SEQ; i += 256) alpha[(size_t)b * SEQ + i] = sc[i] * inv;
}

// ---------------------------------------------------------------------------
__global__ void k_zero(float* __restrict__ p, int n) {
    const int i = blockIdx.x * blockDim.x + threadIdx.x;
    if (i < n) p[i] = 0.f;
}

// context[b,u] = sum_s alpha[b,s] * hidden[b,s,u]  (memory bound: 256 MB)
__global__ void k_context(const float* __restrict__ hidden, const float* __restrict__ alpha,
                          float* __restrict__ ctx) {
    __shared__ float aL[256];
    const int b   = blockIdx.x;        // 32
    const int uc  = blockIdx.y;        // 4
    const int scn = blockIdx.z;        // 8
    const int tid = threadIdx.x;       // 256
    const int u  = uc * 256 + tid;
    const int s0 = scn * 256;
    aL[tid] = alpha[(size_t)b * SEQ + s0 + tid];
    __syncthreads();
    const float* hp = hidden + ((size_t)b * SEQ + s0) * UNITS + u;
    float acc = 0.f;
    #pragma unroll 4
    for (int s = 0; s < 256; ++s)
        acc = fmaf(aL[s], hp[(size_t)s * UNITS], acc);
    atomicAdd(&ctx[(size_t)b * UNITS + u], acc);
}

// ---------------------------------------------------------------------------
extern "C" void kernel_launch(void* const* d_in, const int* in_sizes, int n_in,
                              void* d_out, int out_size, void* d_ws, size_t ws_size,
                              hipStream_t stream) {
    const float* s_prev = (const float*)d_in[0];
    const float* hidden = (const float*)d_in[1];
    const float* Ww     = (const float*)d_in[2];
    const float* Wb     = (const float*)d_in[3];
    const float* Uw     = (const float*)d_in[4];
    const float* Ub     = (const float*)d_in[5];
    const float* Vw     = (const float*)d_in[6];
    const float* Vb     = (const float*)d_in[7];

    float* ctx_out   = (float*)d_out;                    // (32, 1024)
    float* alpha_out = ctx_out + BATCH * UNITS;          // (32, 2048, 1)

    char* ws = (char*)d_ws;
    __bf16* Ut   = (__bf16*)ws;                                        // 2 MB
    float*  dec  = (float*)(ws + 2u * 1024u * 1024u);                  // 128 KB
    float*  scrs = (float*)(ws + 2u * 1024u * 1024u + 128u * 1024u);   // 256 KB

    k_transpose_uw<<<dim3(UNITS / 32, UNITS / 32), dim3(32, 8), 0, stream>>>(Uw, Ut);
    k_decoder<<<(BATCH * UNITS) / 256, 256, 0, stream>>>(s_prev, Ww, Wb, dec);

    const size_t smem = (size_t)MTILE * APAD * 2 + 2 * UNITS * 4;  // 272,384 B (<320 KB/WGP)
    k_enc_scores<<<(BATCH * SEQ) / MTILE, 256, smem, stream>>>(hidden, Ut, dec, Ub, Vw, Vb, scrs);

    k_softmax<<<BATCH, 256, 0, stream>>>(scrs, alpha_out);
    k_zero<<<(BATCH * UNITS) / 256, 256, 0, stream>>>(ctx_out, BATCH * UNITS);
    k_context<<<dim3(BATCH, 4, 8), 256, 0, stream>>>(hidden, alpha_out, ctx_out);
}
